// VectorQuantizer_14843406975525
// MI455X (gfx1250) — matrix operand order repaired and checked by reading, not compiled, not run on hardware
//
#include <hip/hip_runtime.h>

// VQ-VAE vector quantizer for MI455X (gfx1250, wave32).
//   z   : [8, 64, 16, 32, 32] fp32   (d_in[0])
//   emb : [512, 64] fp32             (d_in[1])
//   d_out = [ out (8*64*16*32*32 fp32) | loss (1 fp32) | indices (131072 fp32) ]
//
// GEMM Z(131072x64) x E^T(64x512) on v_wmma_f32_16x16x32_f16 (f16 in, f32 acc),
// argmin over 512 codes per row, exact fp32 gather + loss.
// Round 2: 2-way software pipelined WMMA inner loop (two independent
// accumulators) + branchless u64-key argmin reduction.

typedef __attribute__((ext_vector_type(16))) _Float16 v16h;
typedef __attribute__((ext_vector_type(8)))  float    v8f;

#define NROWS   131072      // B*D*H*W
#define CH      64
#define KCODES  512
#define QPB     16384       // D*H*W (rows per batch image)
#define BM      128         // rows per block
#define NBLK    (NROWS / BM)   // 1024

// LDS layout (dynamic): fp32 z tile [128][65], f16 emb [512][80],
// enorm [512] f32, winner idx [128] i32.
#define A32_STRIDE   65
#define EH_STRIDE    80     // halves; 160B rows -> every frag 32B aligned
#define OFF_EH       (BM * A32_STRIDE * 4)                 // 33280
#define OFF_ENORM    (OFF_EH + KCODES * EH_STRIDE * 2)     // +81920
#define OFF_SIDX     (OFF_ENORM + KCODES * 4)              // +2048
#define SMEM_BYTES   (OFF_SIDX + BM * 4)                   // 117,760 B

__device__ __forceinline__ v8f wmma_f16(v16h a, v16h b, v8f c) {
    return __builtin_amdgcn_wmma_f32_16x16x32_f16(
        false, a, false, b, (short)0, c, false, false);
}

__global__ __launch_bounds__(256) void vq_main_kernel(
    const float* __restrict__ z,
    const float* __restrict__ emb,
    float* __restrict__ out,
    float* __restrict__ idx_out,
    float* __restrict__ block_sums)
{
    extern __shared__ char smem[];
    float*    a32   = (float*)smem;                // [128][65] fp32 z tile
    _Float16* eh    = (_Float16*)(smem + OFF_EH);  // [512][80] f16 codebook
    float*    enorm = (float*)(smem + OFF_ENORM);  // [512] ||e_k||^2
    int*      s_idx = (int*)(smem + OFF_SIDX);     // [128] winners

    const int tid   = threadIdx.x;
    const int blk   = blockIdx.x;
    const int n0    = blk * BM;
    const int bimg  = n0 >> 14;          // n0 / QPB
    const int q0    = n0 & (QPB - 1);
    const float* zb = z + (size_t)bimg * (CH * QPB);

    // ---- stage fp32 z tile: a32[i][c] = z[b, c, (q0+i)] ; coalesced in i ----
    #pragma unroll 4
    for (int t = 0; t < 32; ++t) {
        int v = t * 256 + tid;
        int c = v >> 7;           // 0..63
        int i = v & 127;          // 0..127
        a32[i * A32_STRIDE + c] = zb[c * QPB + q0 + i];
    }

    // ---- stage codebook as f16 (float4 coalesced reads) ----
    const float4* emb4 = (const float4*)emb;   // 512 rows x 16 float4
    #pragma unroll 4
    for (int t = 0; t < 32; ++t) {
        int v = t * 256 + tid;    // 0..8191
        int k = v >> 4;
        int j = v & 15;
        float4 e4 = emb4[v];
        _Float16* p = &eh[k * EH_STRIDE + j * 4];
        p[0] = (_Float16)e4.x; p[1] = (_Float16)e4.y;
        p[2] = (_Float16)e4.z; p[3] = (_Float16)e4.w;
    }

    // ---- ||e_k||^2 in fp32 ----
    for (int k = tid; k < KCODES; k += 256) {
        const float* er = emb + k * CH;
        float s = 0.f;
        #pragma unroll
        for (int c = 0; c < CH; ++c) s = fmaf(er[c], er[c], s);
        enorm[k] = s;
    }
    __syncthreads();

    // ---- per-wave 16-row M subtile; sweep 32 code subtiles of 16 ----
    const int lane = tid & 31;
    const int wave = tid >> 5;     // 0..7 -> M subtile
    const int lh   = lane >> 4;    // half-wave
    const int ln   = lane & 15;

    // A fragments (contiguous k order, matching B):
    //   lane half lh, element e:  c = kc*32 + 16*lh + e         (e < 16)
    const int arow = wave * 16 + ln;
    v16h afrag[2];
    #pragma unroll
    for (int kc = 0; kc < 2; ++kc) {
        const float* ap = &a32[arow * A32_STRIDE + kc * 32 + 16 * lh];
        v16h a;
        #pragma unroll
        for (int e = 0; e < 8; ++e) a[e]     = (_Float16)ap[e];
        #pragma unroll
        for (int e = 0; e < 8; ++e) a[8 + e] = (_Float16)ap[8 + e];
        afrag[kc] = a;
    }

    float bestv[8];
    int   bestk[8];
    #pragma unroll
    for (int r = 0; r < 8; ++r) { bestv[r] = 3.4e38f; bestk[r] = 0x7fffffff; }

    // Two code subtiles per iteration: independent accumulators so the
    // score processing of tile A overlaps the WMMAs of tile B.
    for (int t = 0; t < 32; t += 2) {
        // B fragments (32K x 16 codes): lane ln = code, elems 0..15 ->
        // c = kc*32 + 16*lh + e  (rows 32B aligned -> ds_load_b128 pairs)
        const _Float16* bpA = &eh[((t + 0) * 16 + ln) * EH_STRIDE + 16 * lh];
        const _Float16* bpB = &eh[((t + 1) * 16 + ln) * EH_STRIDE + 16 * lh];
        v16h b0A = *(const v16h*)(bpA);
        v16h b1A = *(const v16h*)(bpA + 32);
        v16h b0B = *(const v16h*)(bpB);
        v16h b1B = *(const v16h*)(bpB + 32);

        v8f accA = {};
        v8f accB = {};
        accA = wmma_f16(afrag[0], b0A, accA);
        accB = wmma_f16(afrag[0], b0B, accB);
        accA = wmma_f16(afrag[1], b1A, accA);
        accB = wmma_f16(afrag[1], b1B, accB);

        const int   kA = (t + 0) * 16 + ln;
        const int   kB = (t + 1) * 16 + ln;
        const float eA = enorm[kA];
        const float eB = enorm[kB];
        #pragma unroll
        for (int r = 0; r < 8; ++r) {
            float sA = fmaf(-2.0f, accA[r], eA);   // ||e||^2 - 2 z.e
            if (sA < bestv[r]) { bestv[r] = sA; bestk[r] = kA; }
        }
        #pragma unroll
        for (int r = 0; r < 8; ++r) {
            float sB = fmaf(-2.0f, accB[r], eB);
            if (sB < bestv[r]) { bestv[r] = sB; bestk[r] = kB; }
        }
    }

    // ---- branchless argmin across the 16 lanes of each half-wave ----
    // Key = order-preserving float bits (high 32) | index (low 32):
    // unsigned min == smaller score, ties -> smaller index (matches argmin).
    #pragma unroll
    for (int r = 0; r < 8; ++r) {
        unsigned int u = __float_as_uint(bestv[r]);
        u ^= ((unsigned int)((int)u >> 31)) | 0x80000000u;
        unsigned long long key =
            ((unsigned long long)u << 32) | (unsigned int)bestk[r];
        #pragma unroll
        for (int m = 1; m < 16; m <<= 1) {
            unsigned long long o = __shfl_xor(key, m, 16);
            key = (o < key) ? o : key;
        }
        if (ln == 0)
            s_idx[wave * 16 + lh * 8 + r] = (int)(key & 0xffffffffu);  // M = r + 8*lh
    }
    __syncthreads();

    // ---- exact fp32 gather + transposed store + loss partial ----
    float lsum = 0.f;
    #pragma unroll 4
    for (int t = 0; t < 32; ++t) {
        int v = t * 256 + tid;
        int c = v >> 7;
        int i = v & 127;
        int k = s_idx[i];
        float e  = emb[k * CH + c];              // L2-resident gather
        float zv = a32[i * A32_STRIDE + c];
        out[(size_t)bimg * (CH * QPB) + c * QPB + q0 + i] = e;  // quantized_st == quantized
        float d = e - zv;
        lsum = fmaf(d, d, lsum);
    }
    if (tid < BM) idx_out[n0 + tid] = (float)s_idx[tid];

    // deterministic block tree reduction (reuse enorm region)
    float* red = enorm;
    red[tid] = lsum;
    __syncthreads();
    #pragma unroll
    for (int s = 128; s > 0; s >>= 1) {
        if (tid < s) red[tid] += red[tid + s];
        __syncthreads();
    }
    if (tid == 0) block_sums[blk] = red[0];
}

__global__ __launch_bounds__(256) void vq_loss_reduce_kernel(
    const float* __restrict__ block_sums, float* __restrict__ loss_out)
{
    __shared__ float red[256];
    float s = 0.f;
    for (int i = threadIdx.x; i < NBLK; i += 256) s += block_sums[i];
    red[threadIdx.x] = s;
    __syncthreads();
    #pragma unroll
    for (int k = 128; k > 0; k >>= 1) {
        if (threadIdx.x < k) red[threadIdx.x] += red[threadIdx.x + k];
        __syncthreads();
    }
    if (threadIdx.x == 0)
        loss_out[0] = 1.25f * red[0] / (float)(NROWS * CH);  // q_loss + 0.25*e_loss
}

extern "C" void kernel_launch(void* const* d_in, const int* in_sizes, int n_in,
                              void* d_out, int out_size, void* d_ws, size_t ws_size,
                              hipStream_t stream)
{
    const float* z   = (const float*)d_in[0];
    const float* emb = (const float*)d_in[1];

    float* out      = (float*)d_out;                 // 8,388,608 elements
    float* loss     = out + (size_t)NROWS * CH;      // 1 element
    float* idx_out  = loss + 1;                      // 131,072 elements (as float)
    float* blk_sums = (float*)d_ws;                  // 1024 floats of scratch

    vq_main_kernel<<<dim3(NBLK), dim3(256), SMEM_BYTES, stream>>>(
        z, emb, out, idx_out, blk_sums);
    vq_loss_reduce_kernel<<<dim3(1), dim3(256), 0, stream>>>(blk_sums, loss);
}
